// LSTMController_83339545412019
// MI455X (gfx1250) — matrix operand order repaired and checked by reading, not compiled, not run on hardware
//
#include <hip/hip_runtime.h>
#include <hip/hip_bf16.h>

// ---------- vector types ----------
typedef __attribute__((ext_vector_type(16))) __bf16 v16bf;
typedef __attribute__((ext_vector_type(8)))  __bf16 v8bf;
typedef __attribute__((ext_vector_type(4)))  __bf16 v4bf;
typedef __attribute__((ext_vector_type(8)))  float  v8f;

#define H 2048
#define GATE4H 8192
#define NACT 64
#define NSTEPS 32

// ---------- fp32 -> bf16 conversion (one pass; makes weights L2-resident) ----------
__global__ void __launch_bounds__(256)
lstm_cvt_bf16_kernel(const float4* __restrict__ in, v4bf* __restrict__ out, int n4) {
    int i = blockIdx.x * blockDim.x + threadIdx.x;
    int stride = gridDim.x * blockDim.x;
    for (; i < n4; i += stride) {
        float4 v = in[i];
        v4bf o;
        o[0] = (__bf16)v.x; o[1] = (__bf16)v.y;
        o[2] = (__bf16)v.z; o[3] = (__bf16)v.w;
        out[i] = o;
    }
}

// ---------- state init: h=c=0, x=start_token ----------
__global__ void __launch_bounds__(256)
lstm_init_kernel(const float* __restrict__ start, float* __restrict__ h,
                 float* __restrict__ c, __bf16* __restrict__ hb,
                 float* __restrict__ x, __bf16* __restrict__ xb) {
    int i = blockIdx.x * blockDim.x + threadIdx.x;
    if (i < H) { float v = start[i]; x[i] = v; xb[i] = (__bf16)v; }
    if (i < 2 * H) { h[i] = 0.0f; c[i] = 0.0f; hb[i] = (__bf16)0.0f; }
}

// ---------- gates GEMV: gates = Wih@x + Whh@h + bih + bhh  (bf16 WMMA, f32 acc) ----------
// 512 row tiles of 16; one wave per tile; 128 blocks x 4 waves.
__global__ void __launch_bounds__(128)
lstm_gates_wmma_kernel(const __bf16* __restrict__ Wih, const __bf16* __restrict__ Whh,
                       const __bf16* __restrict__ xb,  const __bf16* __restrict__ hb,
                       const float* __restrict__ bih,  const float* __restrict__ bhh,
                       float* __restrict__ gates) {
    const int lane = threadIdx.x & 31;
    const int wave = threadIdx.x >> 5;
    const int tile = blockIdx.x * 4 + wave;         // 0..511
    const int row0 = tile * 16;
    const bool hi  = lane >= 16;
    const int  m   = lane & 15;
    const int  aoff = hi ? 16 : 0;                  // contiguous k = 16h + s (matches B)
    const int  boff = hi ? 16 : 0;                  // contiguous k = 16h + s (matches A)

    const __bf16* arow_ih = Wih + (size_t)(row0 + m) * H;
    const __bf16* arow_hh = Whh + (size_t)(row0 + m) * H;

    v8f acc = {};
    #pragma unroll 4
    for (int k = 0; k < H; k += 32) {
        v8bf alo = *(const v8bf*)(arow_ih + k + aoff);
        v8bf ahi = *(const v8bf*)(arow_ih + k + aoff + 8);
        v16bf a, b;
        #pragma unroll
        for (int i = 0; i < 8; ++i) { a[i] = alo[i]; a[i + 8] = ahi[i]; }
        b = *(const v16bf*)(xb + k + boff);          // x broadcast to all 16 columns
        acc = __builtin_amdgcn_wmma_f32_16x16x32_bf16(false, a, false, b,
                                                      (short)0, acc, false, false);
    }
    #pragma unroll 4
    for (int k = 0; k < H; k += 32) {
        v8bf alo = *(const v8bf*)(arow_hh + k + aoff);
        v8bf ahi = *(const v8bf*)(arow_hh + k + aoff + 8);
        v16bf a, b;
        #pragma unroll
        for (int i = 0; i < 8; ++i) { a[i] = alo[i]; a[i + 8] = ahi[i]; }
        b = *(const v16bf*)(hb + k + boff);
        acc = __builtin_amdgcn_wmma_f32_16x16x32_bf16(false, a, false, b,
                                                      (short)0, acc, false, false);
    }
    // D layout: VGPR r => M=r (lanes 0-15) / M=r+8 (lanes 16-31); all N columns equal.
    if (lane == 0 || lane == 16) {
        int r0 = row0 + (hi ? 8 : 0);
        #pragma unroll
        for (int r = 0; r < 8; ++r)
            gates[r0 + r] = acc[r] + bih[r0 + r] + bhh[r0 + r];
    }
}

// ---------- pointwise LSTM cell ----------
__global__ void __launch_bounds__(256)
lstm_cell_kernel(const float* __restrict__ gates, float* __restrict__ h,
                 float* __restrict__ c, __bf16* __restrict__ hb) {
    int j = blockIdx.x * blockDim.x + threadIdx.x;
    if (j >= H) return;
    float ig = gates[j], fg = gates[H + j], gg = gates[2 * H + j], og = gates[3 * H + j];
    float i_ = 1.0f / (1.0f + expf(-ig));
    float f_ = 1.0f / (1.0f + expf(-fg));
    float g_ = tanhf(gg);
    float o_ = 1.0f / (1.0f + expf(-og));
    float cn = f_ * c[j] + i_ * g_;
    float hn = o_ * tanhf(cn);
    c[j] = cn; h[j] = hn; hb[j] = (__bf16)hn;
}

// ---------- head: logits = head_w[t] @ h + head_b[t]  (64x2048, f32) ----------
__global__ void __launch_bounds__(256)
lstm_head_kernel(const float* __restrict__ hw, const float* __restrict__ hbias,
                 const float* __restrict__ h, float* __restrict__ logits) {
    int lane = threadIdx.x & 31, wave = threadIdx.x >> 5;
    for (int rr = 0; rr < 8; ++rr) {
        int r = wave * 8 + rr;                      // 8 waves x 8 rows = 64
        const float* wrow = hw + (size_t)r * H;
        float s = 0.0f;
        for (int k = lane; k < H; k += 32) s += wrow[k] * h[k];
        #pragma unroll
        for (int mm = 16; mm >= 1; mm >>= 1) s += __shfl_xor(s, mm, 32);
        if (lane == 0) logits[r] = s + hbias[r];
    }
}

// ---------- JAX threefry2x32 ----------
__device__ __forceinline__ void tf2x32(unsigned k0, unsigned k1,
                                       unsigned x0, unsigned x1,
                                       unsigned& o0, unsigned& o1) {
    unsigned ks[3] = { k0, k1, k0 ^ k1 ^ 0x1BD11BDAu };
    const int R[8] = { 13, 15, 26, 6, 17, 29, 16, 24 };
    x0 += ks[0]; x1 += ks[1];
    #pragma unroll
    for (int i = 0; i < 5; ++i) {
        const int* r = (i & 1) ? (R + 4) : R;
        #pragma unroll
        for (int j = 0; j < 4; ++j) {
            x0 += x1;
            x1 = (x1 << r[j]) | (x1 >> (32 - r[j]));
            x1 ^= x0;
        }
        x0 += ks[(i + 1) % 3];
        x1 += ks[(i + 2) % 3] + (unsigned)(i + 1);
    }
    o0 = x0; o1 = x1;
}

__device__ __forceinline__ float bits_to_gumbel(unsigned b) {
    const float TINY = 1.1754943508222875e-38f;     // finfo(float32).tiny
    unsigned u = (b >> 9) | 0x3f800000u;
    float f = __uint_as_float(u) - 1.0f;            // uniform in [0,1)
    f = fmaxf(TINY, f * (1.0f - TINY) + TINY);      // uniform in [tiny,1)
    return -logf(-logf(f));
}

// ---------- sampler: gumbel-max categorical + log_softmax + feedback (1 wave) ----------
__global__ void __launch_bounds__(32)
lstm_sample_kernel(const float* __restrict__ logits, const float* __restrict__ action_emb,
                   float* __restrict__ x, __bf16* __restrict__ xb,
                   float* __restrict__ out, float* __restrict__ acc, int t) {
    int lane = threadIdx.x;                          // 0..31
    // key_t = (flat[2t], flat[2t+1]) from jax.random.split(key(42), 32)
    unsigned k0, k1, o0, o1;
    unsigned ut = (unsigned)t;
    if (t < 16) {
        tf2x32(0u, 42u, 2u * ut,      32u + 2u * ut, o0, o1); k0 = o0;
        tf2x32(0u, 42u, 2u * ut + 1u, 33u + 2u * ut, o0, o1); k1 = o0;
    } else {
        tf2x32(0u, 42u, 2u * ut - 32u, 2u * ut,      o0, o1); k0 = o1;
        tf2x32(0u, 42u, 2u * ut - 31u, 2u * ut + 1u, o0, o1); k1 = o1;
    }
    // gumbel bits for 64 categories: lane i covers categories i and i+32
    unsigned b0, b1;
    tf2x32(k0, k1, (unsigned)lane, (unsigned)(lane + 32), b0, b1);
    float l0 = logits[lane], l1 = logits[lane + 32];
    float v0 = l0 + bits_to_gumbel(b0);
    float v1 = l1 + bits_to_gumbel(b1);
    // argmax, tie -> lowest index (matches jnp.argmax)
    float bv = v0; int bi = lane;
    if (v1 > bv) { bv = v1; bi = lane + 32; }
    #pragma unroll
    for (int mm = 16; mm >= 1; mm >>= 1) {
        float ov = __shfl_xor(bv, mm, 32);
        int   oi = __shfl_xor(bi, mm, 32);
        if (ov > bv || (ov == bv && oi < bi)) { bv = ov; bi = oi; }
    }
    // logsumexp over raw logits
    float mx = fmaxf(l0, l1);
    #pragma unroll
    for (int mm = 16; mm >= 1; mm >>= 1) mx = fmaxf(mx, __shfl_xor(mx, mm, 32));
    float s = expf(l0 - mx) + expf(l1 - mx);
    #pragma unroll
    for (int mm = 16; mm >= 1; mm >>= 1) s += __shfl_xor(s, mm, 32);

    int a = bi;                                      // all lanes agree
    float logp = logits[a] - (mx + logf(s));
    if (lane == 0) {
        out[t] = (float)a;
        float na = (t == 0 ? 0.0f : *acc) + logp;
        *acc = na;
        if (t == NSTEPS - 1) out[NSTEPS] = na;       // sum of logps
    }
    // next input = action_emb[a]
    for (int j = lane; j < H; j += 32) {
        float v = action_emb[(size_t)a * H + j];
        x[j] = v; xb[j] = (__bf16)v;
    }
}

extern "C" void kernel_launch(void* const* d_in, const int* in_sizes, int n_in,
                              void* d_out, int out_size, void* d_ws, size_t ws_size,
                              hipStream_t stream) {
    (void)in_sizes; (void)n_in; (void)out_size; (void)ws_size;
    const float* start_token = (const float*)d_in[0];
    const float* action_emb  = (const float*)d_in[1];
    const float* w_ih        = (const float*)d_in[2];
    const float* w_hh        = (const float*)d_in[3];
    const float* b_ih        = (const float*)d_in[4];
    const float* b_hh        = (const float*)d_in[5];
    const float* head_w      = (const float*)d_in[6];
    const float* head_b      = (const float*)d_in[7];
    float* out = (float*)d_out;

    // workspace carve-out (bf16 weights: 2x64MB -> fits the 192MB L2)
    char* ws = (char*)d_ws;
    size_t off = 0;
    auto take = [&](size_t bytes) -> char* {
        char* p = ws + off;
        off = (off + bytes + 255) & ~(size_t)255;
        return p;
    };
    const size_t WELEM = (size_t)2 * GATE4H * H;     // per-tensor weight elements
    __bf16* wihb  = (__bf16*)take(WELEM * 2);
    __bf16* whhb  = (__bf16*)take(WELEM * 2);
    float*  gates = (float*) take((size_t)GATE4H * 4);
    float*  h     = (float*) take((size_t)2 * H * 4);
    float*  c     = (float*) take((size_t)2 * H * 4);
    __bf16* hb    = (__bf16*)take((size_t)2 * H * 2);
    float*  x     = (float*) take((size_t)H * 4);
    __bf16* xb    = (__bf16*)take((size_t)H * 2);
    float*  logits= (float*) take((size_t)NACT * 4);
    float*  acc   = (float*) take(256);

    // one-time (per call) weight downconversion
    lstm_cvt_bf16_kernel<<<2048, 256, 0, stream>>>((const float4*)w_ih, (v4bf*)wihb, (int)(WELEM / 4));
    lstm_cvt_bf16_kernel<<<2048, 256, 0, stream>>>((const float4*)w_hh, (v4bf*)whhb, (int)(WELEM / 4));
    lstm_init_kernel<<<16, 256, 0, stream>>>(start_token, h, c, hb, x, xb);

    for (int t = 0; t < NSTEPS; ++t) {
        for (int l = 0; l < 2; ++l) {
            const __bf16* xin = (l == 0) ? xb : hb;  // layer 1 input = layer 0's new h
            lstm_gates_wmma_kernel<<<128, 128, 0, stream>>>(
                wihb + (size_t)l * GATE4H * H, whhb + (size_t)l * GATE4H * H,
                xin, hb + (size_t)l * H,
                b_ih + (size_t)l * GATE4H, b_hh + (size_t)l * GATE4H, gates);
            lstm_cell_kernel<<<8, 256, 0, stream>>>(gates, h + (size_t)l * H,
                                                    c + (size_t)l * H, hb + (size_t)l * H);
        }
        lstm_head_kernel<<<1, 256, 0, stream>>>(head_w + (size_t)t * NACT * H,
                                                head_b + (size_t)t * NACT,
                                                h + H, logits);
        lstm_sample_kernel<<<1, 32, 0, stream>>>(logits, action_emb, x, xb, out, acc, t);
    }
}
